// OptimizedMultiheadAttention_8014408974978
// MI455X (gfx1250) — compile-verified, matrix operand order checked by reading, not run on hardware
//
#include <hip/hip_runtime.h>
#include <hip/hip_bf16.h>

typedef __attribute__((ext_vector_type(16))) __bf16 v16bf;
typedef __attribute__((ext_vector_type(8)))  __bf16 v8bf;
typedef __attribute__((ext_vector_type(8)))  float  v8f;

#define BATCH 2
#define SEQ   2048
#define DIM   1024
#define HEADS 16
#define HDIM  64
#define NTOK  (BATCH * SEQ)             // 4096
#define QKVN  (3 * DIM)                 // 3072
#define OUT_ELEMS ((size_t)NTOK * DIM)  // attn_weights offset inside d_out

// ---------------------------------------------------------------------------
// CDNA5 WMMA bf16 16x16x32 fragments (wave32), ISA §7.12.2 layout:
// lane L (0..15) holds row M=L, kbase=0; lanes 16..31 hold kbase=8.
// Fragment elements e=0..7  -> k = kbase + e        (contiguous 16B)
//                   e=8..15 -> k = kbase + 16 + e-8 (contiguous 16B)
// => per-lane fragment = two global_load_b128 when operand is K-contiguous.
// B (KxN) mirrors A with N in place of M, so any operand stored [N][K]
// row-major loads identically. C/D: lane L -> col L&15, VGPR r -> row
// r + (L>=16 ? 8 : 0).
// ---------------------------------------------------------------------------
__device__ __forceinline__ v16bf frag_rows_bf16(const __bf16* __restrict__ p) {
  v8bf lo = *(const v8bf*)(p);
  v8bf hi = *(const v8bf*)(p + 16);
  v16bf f;
#pragma unroll
  for (int i = 0; i < 8; ++i) { f[i] = lo[i]; f[i + 8] = hi[i]; }
  return f;
}

__device__ __forceinline__ v16bf frag_rows_f32(const float* __restrict__ p) {
  v16bf f;
#pragma unroll
  for (int i = 0; i < 8; ++i) { f[i] = (__bf16)p[i]; f[i + 8] = (__bf16)p[16 + i]; }
  return f;
}

#define WMMA_BF16(a, b, c) \
  __builtin_amdgcn_wmma_f32_16x16x32_bf16(false, (a), false, (b), (short)0, (c), false, false)

// WGP-scope prefetch (locality 3 -> pulls into all cache levels on miss)
#define PREFETCH_WGP(p) __builtin_prefetch((const void*)(p), 0, 3)

// Per-lane fragment coordinates
#define LANE      (threadIdx.x & 31)
#define FRAG_R    (LANE & 15)
#define FRAG_KB   (((LANE >> 4) & 1) << 3)
// Wave position inside the 2x2 wave grid of a 128-thread block
#define WAVE_M    ((threadIdx.x >> 6) & 1)
#define WAVE_N    ((threadIdx.x >> 5) & 1)

// ---------------------------------------------------------------------------
// Prepass: f32 -> bf16 elementwise (8 elems/thread, b128 in / b128 out)
// ---------------------------------------------------------------------------
__global__ __launch_bounds__(256) void k_cvt_bf16(const float* __restrict__ in,
                                                  __bf16* __restrict__ out) {
  size_t i = (((size_t)blockIdx.x * 256) + threadIdx.x) * 8;
  v8bf o;
#pragma unroll
  for (int j = 0; j < 8; ++j) o[j] = (__bf16)in[i + j];
  *(v8bf*)(out + i) = o;
}

// ---------------------------------------------------------------------------
// Prepass: transpose + convert  W[K][N] f32  ->  Wt[N][K] bf16  (LDS tiled)
// ---------------------------------------------------------------------------
__global__ __launch_bounds__(256) void k_transpose_cvt(const float* __restrict__ in,
                                                       __bf16* __restrict__ out,
                                                       int K, int N) {
  __shared__ float tile[32][33];
  int tx = threadIdx.x & 31, ty = threadIdx.x >> 5;  // 32 x 8
  int k0 = blockIdx.y * 32, n0 = blockIdx.x * 32;
#pragma unroll
  for (int i = 0; i < 32; i += 8)
    tile[ty + i][tx] = in[(size_t)(k0 + ty + i) * N + n0 + tx];
  __syncthreads();
#pragma unroll
  for (int i = 0; i < 32; i += 8)
    out[(size_t)(n0 + ty + i) * K + k0 + tx] = (__bf16)tile[tx][ty + i];
}

// ---------------------------------------------------------------------------
// Kernel 1: qkv = Xb @ Wtq^T + b ; scatter Q,K [B,H,S,Hd] bf16, V [B,H,Hd,S]
// grid: (QKVN/64, NTOK/64), 128 thr (2x2 waves, each 32x32 tile)
// ---------------------------------------------------------------------------
__global__ __launch_bounds__(128) void k_qkv(const __bf16* __restrict__ Xb,
                                             const __bf16* __restrict__ Wt,
                                             const float* __restrict__ bias,
                                             __bf16* __restrict__ Qb,
                                             __bf16* __restrict__ Kb,
                                             __bf16* __restrict__ Vt) {
  int row0 = blockIdx.y * 64 + WAVE_M * 32;
  int col0 = blockIdx.x * 64 + WAVE_N * 32;
  const __bf16* ap = Xb + (size_t)(row0 + FRAG_R) * DIM + FRAG_KB;
  const __bf16* bp = Wt + (size_t)(col0 + FRAG_R) * DIM + FRAG_KB;

  v8f c00 = {}, c01 = {}, c10 = {}, c11 = {};
  for (int k = 0; k < DIM; k += 32) {
    PREFETCH_WGP(ap + k + 512);
    PREFETCH_WGP(bp + k + 512);
    v16bf a0 = frag_rows_bf16(ap + k);
    v16bf a1 = frag_rows_bf16(ap + k + (size_t)16 * DIM);
    v16bf b0 = frag_rows_bf16(bp + k);
    v16bf b1 = frag_rows_bf16(bp + k + (size_t)16 * DIM);
    c00 = WMMA_BF16(a0, b0, c00);
    c01 = WMMA_BF16(a0, b1, c01);
    c10 = WMMA_BF16(a1, b0, c10);
    c11 = WMMA_BF16(a1, b1, c11);
  }

  int n = LANE & 15, mg = ((LANE >> 4) & 1) << 3;
  const v8f* accs[4] = {&c00, &c01, &c10, &c11};
#pragma unroll
  for (int t = 0; t < 4; ++t) {
    int rb = row0 + (t >> 1) * 16;
    int col = col0 + (t & 1) * 16 + n;
    int which = col >> 10, d = col & (DIM - 1);
    int h = d >> 6, hd = d & (HDIM - 1);
    float bv = bias[col];
    const v8f& c = *accs[t];
#pragma unroll
    for (int r = 0; r < 8; ++r) {
      int row = rb + mg + r;
      int bb = row >> 11, s = row & (SEQ - 1);
      int bh = bb * HEADS + h;
      __bf16 val = (__bf16)(c[r] + bv);
      if (which == 0)      Qb[(((size_t)bh * SEQ) + s) * HDIM + hd] = val;
      else if (which == 1) Kb[(((size_t)bh * SEQ) + s) * HDIM + hd] = val;
      else                 Vt[(((size_t)bh * HDIM) + hd) * SEQ + s] = val;
    }
  }
}

// ---------------------------------------------------------------------------
// Kernel 2: scores = (Q @ K^T) * 0.125 -> f32 d_out attn region
// grid: (SEQ/64, SEQ/64, B*H)
// ---------------------------------------------------------------------------
__global__ __launch_bounds__(128) void k_scores(const __bf16* __restrict__ Qb,
                                                const __bf16* __restrict__ Kb,
                                                float* __restrict__ attn) {
  int bh = blockIdx.z;
  int row0 = blockIdx.y * 64 + WAVE_M * 32;
  int col0 = blockIdx.x * 64 + WAVE_N * 32;
  const __bf16* ap = Qb + ((size_t)bh * SEQ + row0 + FRAG_R) * HDIM + FRAG_KB;
  const __bf16* bp = Kb + ((size_t)bh * SEQ + col0 + FRAG_R) * HDIM + FRAG_KB;

  v8f c00 = {}, c01 = {}, c10 = {}, c11 = {};
#pragma unroll
  for (int k = 0; k < HDIM; k += 32) {
    v16bf a0 = frag_rows_bf16(ap + k);
    v16bf a1 = frag_rows_bf16(ap + k + 16 * HDIM);
    v16bf b0 = frag_rows_bf16(bp + k);
    v16bf b1 = frag_rows_bf16(bp + k + 16 * HDIM);
    c00 = WMMA_BF16(a0, b0, c00);
    c01 = WMMA_BF16(a0, b1, c01);
    c10 = WMMA_BF16(a1, b0, c10);
    c11 = WMMA_BF16(a1, b1, c11);
  }

  int n = LANE & 15, mg = ((LANE >> 4) & 1) << 3;
  float* out = attn + ((size_t)bh * SEQ + row0) * SEQ + col0;
  const v8f* accs[4] = {&c00, &c01, &c10, &c11};
#pragma unroll
  for (int t = 0; t < 4; ++t) {
    float* o = out + (size_t)((t >> 1) * 16) * SEQ + (t & 1) * 16;
    const v8f& c = *accs[t];
#pragma unroll
    for (int r = 0; r < 8; ++r)
      o[(size_t)(mg + r) * SEQ + n] = c[r] * 0.125f;
  }
}

// ---------------------------------------------------------------------------
// Kernel 3: row softmax in place (one block per row, LDS tree reduction)
// ---------------------------------------------------------------------------
__global__ __launch_bounds__(256) void k_softmax(float* __restrict__ attn) {
  __shared__ float red[256];
  float* p = attn + (size_t)blockIdx.x * SEQ;
  int t = threadIdx.x;

  float vals[8];
  float m = -__FLT_MAX__;
#pragma unroll
  for (int i = 0; i < 8; ++i) {
    vals[i] = p[t + i * 256];
    m = fmaxf(m, vals[i]);
  }
  red[t] = m;
  __syncthreads();
  for (int s = 128; s > 0; s >>= 1) {
    if (t < s) red[t] = fmaxf(red[t], red[t + s]);
    __syncthreads();
  }
  m = red[0];
  __syncthreads();

  float sum = 0.0f;
#pragma unroll
  for (int i = 0; i < 8; ++i) {
    vals[i] = __expf(vals[i] - m);
    sum += vals[i];
  }
  red[t] = sum;
  __syncthreads();
  for (int s = 128; s > 0; s >>= 1) {
    if (t < s) red[t] += red[t + s];
    __syncthreads();
  }
  float inv = 1.0f / red[0];
#pragma unroll
  for (int i = 0; i < 8; ++i) p[t + i * 256] = vals[i] * inv;
}

// ---------------------------------------------------------------------------
// Kernel 4: attn_out = P @ V  (P f32 -> bf16 in-reg; V pre-transposed [Hd][S])
// grid: (1, SEQ/64, B*H); writes bf16 AO in [B,S,D]
// ---------------------------------------------------------------------------
__global__ __launch_bounds__(128) void k_av(const float* __restrict__ attn,
                                            const __bf16* __restrict__ Vt,
                                            __bf16* __restrict__ AO) {
  int bh = blockIdx.z;
  int row0 = blockIdx.y * 64 + WAVE_M * 32;
  int col0 = WAVE_N * 32;  // HDIM = 64 = one block in N
  const float*  ap = attn + ((size_t)bh * SEQ + row0 + FRAG_R) * SEQ + FRAG_KB;
  const __bf16* bp = Vt + ((size_t)bh * HDIM + col0 + FRAG_R) * SEQ + FRAG_KB;

  v8f c00 = {}, c01 = {}, c10 = {}, c11 = {};
  for (int k = 0; k < SEQ; k += 32) {
    PREFETCH_WGP(ap + k + 512);
    PREFETCH_WGP(bp + k + 512);
    v16bf a0 = frag_rows_f32(ap + k);
    v16bf a1 = frag_rows_f32(ap + k + (size_t)16 * SEQ);
    v16bf b0 = frag_rows_bf16(bp + k);
    v16bf b1 = frag_rows_bf16(bp + k + (size_t)16 * SEQ);
    c00 = WMMA_BF16(a0, b0, c00);
    c01 = WMMA_BF16(a0, b1, c01);
    c10 = WMMA_BF16(a1, b0, c10);
    c11 = WMMA_BF16(a1, b1, c11);
  }

  int n = LANE & 15, mg = ((LANE >> 4) & 1) << 3;
  int bb = bh >> 4, h = bh & (HEADS - 1);
  const v8f* accs[4] = {&c00, &c01, &c10, &c11};
#pragma unroll
  for (int t = 0; t < 4; ++t) {
    int d = h * HDIM + col0 + (t & 1) * 16 + n;
    int rb = row0 + (t >> 1) * 16;
    const v8f& c = *accs[t];
#pragma unroll
    for (int r = 0; r < 8; ++r) {
      int s = rb + mg + r;
      AO[((size_t)(bb * SEQ + s)) * DIM + d] = (__bf16)c[r];
    }
  }
}

// ---------------------------------------------------------------------------
// Kernel 5: out = AO @ Wto^T + b_out -> f32 ; grid: (DIM/64, NTOK/64)
// ---------------------------------------------------------------------------
__global__ __launch_bounds__(128) void k_out(const __bf16* __restrict__ AO,
                                             const __bf16* __restrict__ Wt,
                                             const float* __restrict__ bias,
                                             float* __restrict__ Y) {
  int row0 = blockIdx.y * 64 + WAVE_M * 32;
  int col0 = blockIdx.x * 64 + WAVE_N * 32;
  const __bf16* ap = AO + (size_t)(row0 + FRAG_R) * DIM + FRAG_KB;
  const __bf16* bp = Wt + (size_t)(col0 + FRAG_R) * DIM + FRAG_KB;

  v8f c00 = {}, c01 = {}, c10 = {}, c11 = {};
  for (int k = 0; k < DIM; k += 32) {
    PREFETCH_WGP(ap + k + 512);
    PREFETCH_WGP(bp + k + 512);
    v16bf a0 = frag_rows_bf16(ap + k);
    v16bf a1 = frag_rows_bf16(ap + k + (size_t)16 * DIM);
    v16bf b0 = frag_rows_bf16(bp + k);
    v16bf b1 = frag_rows_bf16(bp + k + (size_t)16 * DIM);
    c00 = WMMA_BF16(a0, b0, c00);
    c01 = WMMA_BF16(a0, b1, c01);
    c10 = WMMA_BF16(a1, b0, c10);
    c11 = WMMA_BF16(a1, b1, c11);
  }

  int n = LANE & 15, mg = ((LANE >> 4) & 1) << 3;
  const v8f* accs[4] = {&c00, &c01, &c10, &c11};
#pragma unroll
  for (int t = 0; t < 4; ++t) {
    int col = col0 + (t & 1) * 16 + n;
    int rb = row0 + (t >> 1) * 16;
    float bv = bias[col];
    const v8f& c = *accs[t];
#pragma unroll
    for (int r = 0; r < 8; ++r)
      Y[(size_t)(rb + mg + r) * DIM + col] = c[r] + bv;
  }
}

// ---------------------------------------------------------------------------
extern "C" void kernel_launch(void* const* d_in, const int* in_sizes, int n_in,
                              void* d_out, int out_size, void* d_ws, size_t ws_size,
                              hipStream_t stream) {
  const float* X    = (const float*)d_in[0];  // [B,S,D]
  const float* Wqkv = (const float*)d_in[1];  // [D, 3D]
  const float* bqkv = (const float*)d_in[2];  // [3D]
  const float* Wout = (const float*)d_in[3];  // [D, D]
  const float* bout = (const float*)d_in[4];  // [D]

  float* out  = (float*)d_out;       // [B,S,D]
  float* attn = out + OUT_ELEMS;     // [B,H,S,S]

  // Workspace layout (bf16): Xb 8MB, Wtq 6MB, Wto 2MB, Qb 8MB, Kb 8MB,
  // Vt 8MB, AO 8MB = 48 MB total
  __bf16* Xb  = (__bf16*)d_ws;
  __bf16* Wtq = Xb  + (size_t)NTOK * DIM;     // [3072][1024]
  __bf16* Wto = Wtq + (size_t)QKVN * DIM;     // [1024][1024]
  __bf16* Qb  = Wto + (size_t)DIM * DIM;      // [B,H,S,Hd]
  __bf16* Kb  = Qb  + (size_t)NTOK * DIM;     // [B,H,S,Hd]
  __bf16* Vt  = Kb  + (size_t)NTOK * DIM;     // [B,H,Hd,S]
  __bf16* AO  = Vt  + (size_t)NTOK * DIM;     // [B,S,D]

  // Prepasses: convert / transpose operands to K-contiguous bf16
  k_cvt_bf16<<<(NTOK * DIM) / (256 * 8), 256, 0, stream>>>(X, Xb);
  k_transpose_cvt<<<dim3(QKVN / 32, DIM / 32), 256, 0, stream>>>(Wqkv, Wtq, DIM, QKVN);
  k_transpose_cvt<<<dim3(DIM / 32, DIM / 32), 256, 0, stream>>>(Wout, Wto, DIM, DIM);

  // 1) QKV projection
  k_qkv<<<dim3(QKVN / 64, NTOK / 64), 128, 0, stream>>>(Xb, Wtq, bqkv, Qb, Kb, Vt);
  // 2) scores
  k_scores<<<dim3(SEQ / 64, SEQ / 64, BATCH * HEADS), 128, 0, stream>>>(Qb, Kb, attn);
  // 3) softmax
  k_softmax<<<BATCH * HEADS * SEQ, 256, 0, stream>>>(attn);
  // 4) P @ V
  k_av<<<dim3(1, SEQ / 64, BATCH * HEADS), 128, 0, stream>>>(attn, Vt, AO);
  // 5) output projection
  k_out<<<dim3(DIM / 64, NTOK / 64), 128, 0, stream>>>(AO, Wto, bout, out);
}
